// SynthesisLayer_54125177864471
// MI455X (gfx1250) — matrix operand order repaired and checked by reading, not compiled, not run on hardware
//
#include <hip/hip_runtime.h>

// ---------------------------------------------------------------------------
// SynthesisLayer for MI455X (gfx1250, wave32, WMMA bf16)
// out[b] = lrelu( Wn[b] @ x[b] + bias ) * sqrt(2)
// Wn[b]  = rownorm( weight * (1 + lowrank_mod(w[b])) )
// ---------------------------------------------------------------------------

typedef __attribute__((ext_vector_type(16))) __bf16 v16bf;
typedef __attribute__((ext_vector_type(8)))  float  v8f;

#define B_      16
#define CIN     128
#define COUT    128
#define NPTS    65536
#define WDIM    512
#define MODN    768          // (CIN+COUT)*RANK
#define NCHUNK  128          // columns staged per LDS pass
#define CHUNKS  8            // chunks per workgroup
#define XSTRIDE 136          // padded bf16 row stride of transposed x tile

// ============================================================================
// Kernel 1: modulation + demodulation -> per-batch bf16 weight [B][COUT][CIN]
// One block per batch, 128 threads.
// ============================================================================
__global__ __launch_bounds__(128)
void synth_prep_kernel(const float* __restrict__ w,
                       const float* __restrict__ weight,
                       const float* __restrict__ aweight,
                       const float* __restrict__ abias,
                       __bf16* __restrict__ Wn)
{
    __shared__ __align__(16) float w_sh[WDIM];
    __shared__ float mod_sh[MODN];

    const int b = blockIdx.x;
    const int t = threadIdx.x;

    for (int j = t; j < WDIM; j += 128) w_sh[j] = w[b * WDIM + j];
    __syncthreads();

    const float wg = 0.04419417382415922f;   // 1/sqrt(512)

    // mod_params[j] = (w[b] . aweight[j]) * wg + abias[j]
    for (int j = t; j < MODN; j += 128) {
        const float4* a4 = (const float4*)(aweight + (size_t)j * WDIM);
        const float4* w4 = (const float4*)w_sh;
        float acc = 0.f;
        #pragma unroll 4
        for (int d = 0; d < WDIM / 4; ++d) {
            float4 av = a4[d];
            float4 wv = w4[d];
            acc += av.x * wv.x + av.y * wv.y + av.z * wv.z + av.w * wv.w;
        }
        mod_sh[j] = acc * wg + abias[j];
    }
    __syncthreads();

    // thread t owns output row o = t
    const int o = t;
    const float l0 = mod_sh[o * 3 + 0];
    const float l1 = mod_sh[o * 3 + 1];
    const float l2 = mod_sh[o * 3 + 2];
    const float rs3 = 0.5773502691896258f;   // 1/sqrt(3)
    const float* wrow = weight + (size_t)o * CIN;

    float sumsq = 0.f;
    #pragma unroll 4
    for (int i = 0; i < CIN; ++i) {
        float m  = (l0 * mod_sh[384 + i] + l1 * mod_sh[512 + i] + l2 * mod_sh[640 + i]) * rs3;
        float Wv = wrow[i] * (m + 1.f);
        sumsq += Wv * Wv;
    }
    const float inv = 1.f / (sqrtf(sumsq) + 1e-8f);

    __bf16* orow = Wn + ((size_t)b * COUT + o) * CIN;
    #pragma unroll 4
    for (int i = 0; i < CIN; ++i) {
        float m  = (l0 * mod_sh[384 + i] + l1 * mod_sh[512 + i] + l2 * mod_sh[640 + i]) * rs3;
        float Wv = wrow[i] * (m + 1.f) * inv;
        orow[i] = (__bf16)Wv;
    }
}

// ============================================================================
// Kernel 2: batched GEMM  out[b] = Wn[b] @ x[b], fused bias+lrelu*sqrt(2).
// Grid (B, NPTS/(NCHUNK*CHUNKS)), 256 threads = 8 waves.
// Wave w owns M-tile rows [16w, 16w+16); A-fragments resident in VGPRs.
// ============================================================================
__global__ __launch_bounds__(256)
void synth_gemm_kernel(const __bf16* __restrict__ Wn,
                       const float*  __restrict__ x,
                       const float*  __restrict__ bias,
                       float*        __restrict__ out)
{
    __shared__ __align__(16) __bf16 sW[COUT * CIN];       // 32 KB
    __shared__ __align__(16) __bf16 sX[NCHUNK * XSTRIDE]; // ~34 KB, [n][k] transposed

    const int b    = blockIdx.x;
    const int grp  = blockIdx.y;
    const int tid  = threadIdx.x;
    const int lane = tid & 31;
    const int wave = tid >> 5;

    // ---- stage Wn[b] into LDS (bf16 row-major [o][i]) --------------------
    {
        const uint4* src = (const uint4*)(Wn + (size_t)b * COUT * CIN);
        uint4* dst = (uint4*)sW;
        #pragma unroll
        for (int j = tid; j < COUT * CIN / 8; j += 256) dst[j] = src[j];
    }
    __syncthreads();

    // ---- A fragments (16x32 bf16), per ISA wave32 layout -----------------
    // lane<16 : row M=lane,    VGPR0-3 K=kb+0..7,  VGPR4-7 K=kb+16..23
    // lane>=16: row M=lane-16, VGPR0-3 K=kb+8..15, VGPR4-7 K=kb+24..31
    union V16 { v16bf v; uint4 u[2]; };
    const int half = (lane < 16) ? 0 : 1;
    const int mrow = wave * 16 + (lane & 15);

    V16 afrag[4];
    #pragma unroll
    for (int kc = 0; kc < 4; ++kc) {
        const int base = mrow * CIN + kc * 32 + half * 8;
        afrag[kc].u[0] = *(const uint4*)&sW[base];
        afrag[kc].u[1] = *(const uint4*)&sW[base + 16];
    }

    // per-lane bias values for the 8 accumulator VGPRs (D row = v + half*8)
    float bv[8];
    #pragma unroll
    for (int v = 0; v < 8; ++v) bv[v] = bias[wave * 16 + half * 8 + v];

    const int n_thr = tid & 127;   // column within chunk for staging
    const int phase = tid >> 7;    // 0/1: even/odd k-pairs
    const size_t N = NPTS;
    const float* xb = x   + (size_t)b * CIN  * N;
    float*       ob = out + (size_t)b * COUT * N;

    for (int c = 0; c < CHUNKS; ++c) {
        const int n0 = (grp * CHUNKS + c) * NCHUNK;

        // ---- stage x chunk: f32 global -> bf16 LDS transposed [n][k] ----
        #pragma unroll 4
        for (int p = 0; p < 32; ++p) {
            const int k0 = (phase + p * 2) * 2;           // even k
            const float f0 = xb[(size_t)k0 * N + n0 + n_thr];
            const float f1 = xb[(size_t)(k0 + 1) * N + n0 + n_thr];
            union { __bf16 h[2]; unsigned int u; } pk;
            pk.h[0] = (__bf16)f0;
            pk.h[1] = (__bf16)f1;
            *(unsigned int*)&sX[n_thr * XSTRIDE + k0] = pk.u;
            if (c + 1 < CHUNKS)  // warm L2/L0 for the next chunk
                __builtin_prefetch(&xb[(size_t)k0 * N + n0 + NCHUNK + n_thr], 0, 0);
        }
        __syncthreads();

        // ---- compute 8 n-subtiles of 16 columns -------------------------
        #pragma unroll
        for (int nt = 0; nt < 8; ++nt) {
            const int ncol = nt * 16 + (lane & 15);
            v8f acc = {};
            #pragma unroll
            for (int kc = 0; kc < 4; ++kc) {
                // B fragment (32x16): lane<16 K=0..7,16..23, lane>=16 K=8..15,24..31 of slab
                V16 bfrag;
                const int base = ncol * XSTRIDE + kc * 32 + half * 8;
                bfrag.u[0] = *(const uint4*)&sX[base];
                bfrag.u[1] = *(const uint4*)&sX[base + 16];
                acc = __builtin_amdgcn_wmma_f32_16x16x32_bf16(
                    false, afrag[kc].v, false, bfrag.v,
                    (short)0, acc, false, false);
            }
            // ---- epilogue: bias + lrelu * sqrt(2), store f32 ------------
            const size_t ncg = (size_t)n0 + nt * 16 + (lane & 15);
            #pragma unroll
            for (int v = 0; v < 8; ++v) {
                float o = acc[v] + bv[v];
                o = (o >= 0.f ? o : o * 0.2f) * 1.4142135623730951f;
                ob[(size_t)(wave * 16 + half * 8 + v) * N + ncg] = o;
            }
        }
        __syncthreads();
    }
}

// ============================================================================
extern "C" void kernel_launch(void* const* d_in, const int* in_sizes, int n_in,
                              void* d_out, int out_size, void* d_ws, size_t ws_size,
                              hipStream_t stream)
{
    const float* x       = (const float*)d_in[0];  // [B, CIN, NPTS]
    const float* w       = (const float*)d_in[1];  // [B, WDIM]
    const float* weight  = (const float*)d_in[2];  // [COUT, CIN]
    const float* bias    = (const float*)d_in[3];  // [COUT]
    const float* aweight = (const float*)d_in[4];  // [MODN, WDIM]
    const float* abias   = (const float*)d_in[5];  // [MODN]
    float*       out     = (float*)d_out;          // [B, COUT, NPTS]
    __bf16*      Wn      = (__bf16*)d_ws;          // [B, COUT, CIN] bf16 (512 KB)

    synth_prep_kernel<<<B_, 128, 0, stream>>>(w, weight, aweight, abias, Wn);

    dim3 grid(B_, NPTS / (NCHUNK * CHUNKS));       // (16, 64)
    synth_gemm_kernel<<<grid, 256, 0, stream>>>(Wn, x, bias, out);
}
